// LoRALayer_17343077941983
// MI455X (gfx1250) — matrix operand order repaired and checked by reading, not compiled, not run on hardware
//
#include <hip/hip_runtime.h>

typedef __attribute__((ext_vector_type(16))) __bf16 v16bf;
typedef __attribute__((ext_vector_type(8)))  __bf16 v8bf;
typedef __attribute__((ext_vector_type(8)))  float  v8f;
typedef __attribute__((ext_vector_type(4)))  float  v4f;
typedef __attribute__((ext_vector_type(4)))  unsigned int u32x4;
typedef __attribute__((ext_vector_type(8)))  int   i32x8;
typedef __attribute__((ext_vector_type(4)))  int   i32x4;

#define WB 4
#define WT 2048
#define WH 768
#define WR 64
#define WBT (WB*WT)

__device__ __forceinline__ v8f wmma_bf16(v16bf a, v16bf b, v8f c) {
  // D = A(16x32 bf16) * B(32x16 bf16) + C(16x16 f32)
  return __builtin_amdgcn_wmma_f32_16x16x32_bf16(false, a, false, b, (short)0, c, false, false);
}

// TDM: 2D tile load, bf16 elements (data_size=2B). D# per CDNA5 ISA ch.8.
//   tensor: tensor_d1 rows x tensor_d0 elems, row stride stride0 (elems)
//   tile:   tile_d1 rows x tile_d0 elems -> contiguous rows in LDS
__device__ __forceinline__ void tdm_load_2d_bf16(
    const void* gaddr, unsigned lds_off,
    unsigned tensor_d0, unsigned tensor_d1,
    unsigned long long stride0,
    unsigned tile_d0, unsigned tile_d1)
{
  unsigned long long ga = (unsigned long long)gaddr;
  u32x4 g0;
  g0[0] = 1u;                                              // count=1 (valid user D#)
  g0[1] = lds_off;                                         // LDS byte address
  g0[2] = (unsigned)(ga & 0xFFFFFFFFu);                    // global_addr[31:0]
  g0[3] = (unsigned)((ga >> 32) & 0x01FFFFFFu) | (2u << 30); // addr[56:32] | type=2
  i32x8 g1;
  g1[0] = (int)(1u << 16);                                 // data_size=1 (2 bytes)
  g1[1] = (int)((tensor_d0 & 0xFFFFu) << 16);              // tensor_dim0[15:0]
  g1[2] = (int)(((tensor_d0 >> 16) & 0xFFFFu) | ((tensor_d1 & 0xFFFFu) << 16));
  g1[3] = (int)(((tensor_d1 >> 16) & 0xFFFFu) | ((tile_d0 & 0xFFFFu) << 16));
  g1[4] = (int)(tile_d1 & 0xFFFFu);                        // tile_dim1 | tile_dim2=0
  g1[5] = (int)(unsigned)(stride0 & 0xFFFFFFFFu);          // tensor_dim0_stride lo
  g1[6] = (int)(unsigned)((stride0 >> 32) & 0xFFFFu);      // stride hi | dim1_stride=0
  g1[7] = 0;
  i32x4 gz  = {0,0,0,0};
  i32x8 gz8 = {0,0,0,0,0,0,0,0};
  __builtin_amdgcn_tensor_load_to_lds(g0, g1, gz, gz, gz8, 0);
}

__device__ __forceinline__ unsigned lds_offset_of(const void* p) {
  // LDS aperture keeps the byte offset in addr[31:0]
  return (unsigned)(unsigned long long)p;
}

// ---------------------------------------------------------------------------
// Kernel 1: t_q = x @ A_q, t_v = x @ A_v   (M=8192, N=64, K=768), bf16 out
// ---------------------------------------------------------------------------
__global__ __launch_bounds__(256) void k1_lora_a(
    const float* __restrict__ x, const float* __restrict__ Aq,
    const float* __restrict__ Av, __bf16* __restrict__ tq,
    __bf16* __restrict__ tv)
{
  __shared__ __attribute__((aligned(32))) __bf16 aq_t[64*32]; // [n][k] transposed
  __shared__ __attribute__((aligned(32))) __bf16 av_t[64*32];
  const int t    = threadIdx.x;
  const int lane = t & 31;
  const int wv   = t >> 5;
  const int rt   = blockIdx.x * 8 + wv;        // 16-row tile id (0..511)
  const int mrow = rt*16 + (lane & 15);
  const int lo   = (lane < 16) ? 0 : 16;
  const int khalf= (lane < 16) ? 0 : 16;

  v8f accQ[4] = {}, accV[4] = {};

  for (int k0 = 0; k0 < 24; ++k0) {           // K chunks of 32 over H=768
    __syncthreads();
    { // cooperative transposed stage of A chunks (32 x 64 fp32 -> bf16 [n][k])
      const int k  = t >> 3;                  // 0..31
      const int n0 = (t & 7) * 8;             // 0..56
      const float* pq = Aq + (size_t)(k0*32 + k)*WR + n0;
      const float* pv = Av + (size_t)(k0*32 + k)*WR + n0;
      #pragma unroll
      for (int j = 0; j < 8; ++j) {
        aq_t[(n0 + j)*32 + k] = (__bf16)pq[j];
        av_t[(n0 + j)*32 + k] = (__bf16)pv[j];
      }
    }
    __syncthreads();

    // A fragment from x (fp32 -> bf16): K = {lo..lo+15}
    const float* xr = x + (size_t)mrow*WH + k0*32;
    v4f x0 = *(const v4f*)(xr + lo);
    v4f x1 = *(const v4f*)(xr + lo + 4);
    v4f x2 = *(const v4f*)(xr + lo + 8);
    v4f x3 = *(const v4f*)(xr + lo + 12);
    v16bf a;
    #pragma unroll
    for (int i = 0; i < 4; ++i) {
      a[i]    = (__bf16)x0[i];
      a[4+i]  = (__bf16)x1[i];
      a[8+i]  = (__bf16)x2[i];
      a[12+i] = (__bf16)x3[i];
    }
    #pragma unroll
    for (int nt = 0; nt < 4; ++nt) {
      const int n = nt*16 + (lane & 15);
      v16bf bq = *(const v16bf*)&aq_t[n*32 + khalf];
      v16bf bv = *(const v16bf*)&av_t[n*32 + khalf];
      accQ[nt] = wmma_bf16(a, bq, accQ[nt]);
      accV[nt] = wmma_bf16(a, bv, accV[nt]);
    }
  }
  const int half = lane >> 4;
  #pragma unroll
  for (int nt = 0; nt < 4; ++nt) {
    const int n = nt*16 + (lane & 15);
    #pragma unroll
    for (int r = 0; r < 8; ++r) {
      const int row = rt*16 + r + 8*half;
      tq[(size_t)row*WR + n] = (__bf16)accQ[nt][r];
      tv[(size_t)row*WR + n] = (__bf16)accV[nt][r];
    }
  }
}

// ---------------------------------------------------------------------------
// Kernel 2: q = x + t_q @ B_q ; v = x + t_v @ B_v ; xb = bf16(x)
//           v written TRANSPOSED as vbT[batch][h][t] so k4's TDM tile lands
//           in LDS already in [h][key] order.   (M=8192, N=768, K=64)
// ---------------------------------------------------------------------------
__global__ __launch_bounds__(256) void k2_qv(
    const float* __restrict__ x, const __bf16* __restrict__ tq,
    const __bf16* __restrict__ tv, const float* __restrict__ Bq,
    const float* __restrict__ Bv, __bf16* __restrict__ qb,
    __bf16* __restrict__ vbT, __bf16* __restrict__ xb)
{
  __shared__ __attribute__((aligned(32))) __bf16 bq_t[96*64]; // [n][k]
  __shared__ __attribute__((aligned(32))) __bf16 bv_t[96*64];
  const int t     = threadIdx.x;
  const int lane  = t & 31;
  const int wv    = t >> 5;
  const int rt    = blockIdx.x * 8 + wv;     // row tile
  const int nbase = blockIdx.y * 96;         // column chunk
  { // stage B chunks transposed: k=0..63, n=nbase..nbase+95
    const int k  = t >> 2;                   // 0..63
    const int n0 = (t & 3) * 24;             // 0,24,48,72
    const float* pq = Bq + (size_t)k*WH + nbase + n0;
    const float* pv = Bv + (size_t)k*WH + nbase + n0;
    #pragma unroll
    for (int j = 0; j < 24; ++j) {
      bq_t[(n0 + j)*64 + k] = (__bf16)pq[j];
      bv_t[(n0 + j)*64 + k] = (__bf16)pv[j];
    }
  }
  __syncthreads();

  const int mrow  = rt*16 + (lane & 15);
  const int lo    = (lane < 16) ? 0 : 16;
  const int khalf = (lane < 16) ? 0 : 16;

  v8f accQ[6] = {}, accV[6] = {};
  #pragma unroll
  for (int ks = 0; ks < 2; ++ks) {
    const __bf16* tqr = tq + (size_t)mrow*WR + ks*32;
    v8bf q1 = *(const v8bf*)(tqr + lo);
    v8bf q2 = *(const v8bf*)(tqr + lo + 8);
    v16bf aq = __builtin_shufflevector(q1, q2, 0,1,2,3,4,5,6,7,8,9,10,11,12,13,14,15);
    const __bf16* tvr = tv + (size_t)mrow*WR + ks*32;
    v8bf w1 = *(const v8bf*)(tvr + lo);
    v8bf w2 = *(const v8bf*)(tvr + lo + 8);
    v16bf av = __builtin_shufflevector(w1, w2, 0,1,2,3,4,5,6,7,8,9,10,11,12,13,14,15);
    #pragma unroll
    for (int nt = 0; nt < 6; ++nt) {
      const int n = nt*16 + (lane & 15);
      v16bf bq = *(const v16bf*)&bq_t[n*64 + ks*32 + khalf];
      v16bf bv = *(const v16bf*)&bv_t[n*64 + ks*32 + khalf];
      accQ[nt] = wmma_bf16(aq, bq, accQ[nt]);
      accV[nt] = wmma_bf16(av, bv, accV[nt]);
    }
  }
  const int half  = lane >> 4;
  const int batch = rt >> 7;
  const int t0    = (rt & 127) * 16 + 8*half;   // 8 consecutive t per lane
  #pragma unroll
  for (int nt = 0; nt < 6; ++nt) {
    const int n = nbase + nt*16 + (lane & 15);
    v8bf vt8;
    #pragma unroll
    for (int r = 0; r < 8; ++r) {
      const int row = rt*16 + r + 8*half;
      const float xv = x[(size_t)row*WH + n];
      qb[(size_t)row*WH + n] = (__bf16)(xv + accQ[nt][r]);
      xb[(size_t)row*WH + n] = (__bf16)xv;
      vt8[r] = (__bf16)(xv + accV[nt][r]);
    }
    *(v8bf*)(vbT + ((size_t)(batch*WH + n))*WT + t0) = vt8;
  }
}

// ---------------------------------------------------------------------------
// Kernel 3: S = scale * (Q @ K^T) (+mask) -> fp32 logits in ws, plus online
//           row max m and row sum l.  K tiles delivered by the Tensor Data
//           Mover (double-buffered, DMA of tile i+1 overlaps WMMA of tile i).
// ---------------------------------------------------------------------------
__global__ __launch_bounds__(256) void k3_scores(
    const __bf16* __restrict__ qb, const __bf16* __restrict__ xb,
    const int* __restrict__ maskp, float* __restrict__ Sbuf,
    float* __restrict__ mbuf, float* __restrict__ lbuf)
{
  extern __shared__ __bf16 kt[];              // 2 x (16*768) bf16 = 48KB
  const int KT_ELEMS = 16*WH;
  const int t     = threadIdx.x;
  const int lane  = t & 31;
  const int wv    = t >> 5;
  const int rt    = blockIdx.x * 8 + wv;
  const int batch = blockIdx.x >> 4;          // 128 rows/block, 2048/batch
  const int mrow  = rt*16 + (lane & 15);
  const int lo    = (lane < 16) ? 0 : 16;
  const int khalf = (lane < 16) ? 0 : 16;
  const float SCALE = 0.03608439182435161f;   // 1/sqrt(768)

  v16bf qf[24];                                // Q row tile lives in VGPRs
  #pragma unroll
  for (int k0 = 0; k0 < 24; ++k0) {
    const __bf16* qr = qb + (size_t)mrow*WH + k0*32;
    v8bf r1 = *(const v8bf*)(qr + lo);
    v8bf r2 = *(const v8bf*)(qr + lo + 8);
    qf[k0] = __builtin_shufflevector(r1, r2, 0,1,2,3,4,5,6,7,8,9,10,11,12,13,14,15);
  }

  float m[8], l[8];
  #pragma unroll
  for (int r = 0; r < 8; ++r) { m[r] = -__builtin_inff(); l[r] = 0.0f; }

  const __bf16* kp = xb + (size_t)batch*WT*WH;
  const unsigned lds0 = lds_offset_of(kt);

  if (wv == 0) {  // prime the pipeline: TDM load of tile 0
    tdm_load_2d_bf16(kp, lds0, WH, WT, WH, WH, 16);
  }

  for (int kb = 0; kb < 128; ++kb) {          // 16-key tiles
    if (wv == 0) {
      if (kb + 1 < 128) {                     // DMA next tile, wait for current
        tdm_load_2d_bf16(kp + (size_t)(kb+1)*16*WH,
                         lds0 + ((kb+1)&1)*(unsigned)(KT_ELEMS*2),
                         WH, WT, WH, WH, 16);
        __builtin_amdgcn_s_wait_tensorcnt(1);
      } else {
        __builtin_amdgcn_s_wait_tensorcnt(0);
      }
    }
    __syncthreads();                          // tile kb visible to all waves

    const __bf16* ktb = kt + (kb&1)*KT_ELEMS;
    const int n = lane & 15;                  // key column
    v8f acc = {};
    v16bf bcur = *(const v16bf*)&ktb[n*WH + khalf];
    #pragma unroll
    for (int k0 = 0; k0 < 24; ++k0) {        // one fragment ahead
      const int kn = (k0 < 23) ? (k0 + 1) : 23;
      v16bf bnext = *(const v16bf*)&ktb[n*WH + kn*32 + khalf];
      acc = wmma_bf16(qf[k0], bcur, acc);
      bcur = bnext;
    }

    const int col  = kb*16 + (lane & 15);
    const int mk   = maskp[batch*WT + col];
    const int half = lane >> 4;
    #pragma unroll
    for (int r = 0; r < 8; ++r) {
      float s = (mk > 0) ? acc[r]*SCALE : -__builtin_inff();
      Sbuf[(size_t)(rt*16 + r + 8*half)*WT + col] = s;
      float rmax = s;
      #pragma unroll
      for (int d = 1; d < 16; d <<= 1) rmax = fmaxf(rmax, __shfl_xor(rmax, d, 32));
      const float nm = fmaxf(m[r], rmax);
      float e = __expf(s - nm);
      #pragma unroll
      for (int d = 1; d < 16; d <<= 1) e += __shfl_xor(e, d, 32);
      l[r] = l[r]*__expf(m[r] - nm) + e;
      m[r] = nm;
    }
    __syncthreads();                          // done with buffer before reuse
  }
  if ((lane & 15) == 0) {
    const int half = lane >> 4;
    #pragma unroll
    for (int r = 0; r < 8; ++r) {
      mbuf[rt*16 + r + 8*half] = m[r];
      lbuf[rt*16 + r + 8*half] = l[r];
    }
  }
}

// ---------------------------------------------------------------------------
// Kernel 4: out = softmax(S) @ V  (M=8192, N=768, K=2048 per batch)
//           P = exp(S-m)/l built on the fly as bf16 A-fragment.
//           V tiles (already transposed in memory) streamed by TDM,
//           double-buffered: tile = 768 h-rows x 32 keys -> LDS [h][key].
// ---------------------------------------------------------------------------
__global__ __launch_bounds__(256) void k4_out(
    const float* __restrict__ Sbuf, const float* __restrict__ mbuf,
    const float* __restrict__ lbuf, const __bf16* __restrict__ vbT,
    float* __restrict__ out)
{
  extern __shared__ __bf16 vt[];              // 2 x (768*32) bf16 = 96KB
  const int VT_ELEMS = WH*32;
  const int t     = threadIdx.x;
  const int lane  = t & 31;
  const int wv    = t >> 5;
  const int rt    = blockIdx.x;               // 16-row tile
  const int batch = rt >> 7;
  const int nb    = wv * 96;
  const int row_a = rt*16 + (lane & 15);
  const int lo    = (lane < 16) ? 0 : 16;
  const int khalf = (lane < 16) ? 0 : 16;

  const float mr   = mbuf[row_a];
  const float linv = 1.0f / lbuf[row_a];
  const float* Srow = Sbuf + (size_t)row_a*WT;
  const __bf16* vbase = vbT + (size_t)batch*WH*WT;   // [h][t] for this batch
  const unsigned lds0 = lds_offset_of(vt);

  if (wv == 0) {  // prime: tile 0 = columns [0,32) of the [768][2048] tensor
    tdm_load_2d_bf16(vbase, lds0, WT, WH, WT, 32, WH);
  }

  v8f acc[6] = {};
  for (int ks = 0; ks < 64; ++ks) {           // 32-key steps
    if (wv == 0) {
      if (ks + 1 < 64) {
        tdm_load_2d_bf16(vbase + (size_t)(ks+1)*32,
                         lds0 + ((ks+1)&1)*(unsigned)(VT_ELEMS*2),
                         WT, WH, WT, 32, WH);
        __builtin_amdgcn_s_wait_tensorcnt(1);
      } else {
        __builtin_amdgcn_s_wait_tensorcnt(0);
      }
    }
    __syncthreads();

    const __bf16* vtb = vt + (ks&1)*VT_ELEMS;

    // P fragment: exp(S - m) / l, fp32 -> bf16
    const float* sp = Srow + ks*32;
    v4f s0 = *(const v4f*)(sp + lo);
    v4f s1 = *(const v4f*)(sp + lo + 4);
    v4f s2 = *(const v4f*)(sp + lo + 8);
    v4f s3 = *(const v4f*)(sp + lo + 12);
    v16bf a;
    #pragma unroll
    for (int i = 0; i < 4; ++i) {
      a[i]    = (__bf16)(__expf(s0[i] - mr) * linv);
      a[4+i]  = (__bf16)(__expf(s1[i] - mr) * linv);
      a[8+i]  = (__bf16)(__expf(s2[i] - mr) * linv);
      a[12+i] = (__bf16)(__expf(s3[i] - mr) * linv);
    }
    #pragma unroll
    for (int nt = 0; nt < 6; ++nt) {
      const int n = nb + nt*16 + (lane & 15);
      v16bf bf = *(const v16bf*)&vtb[n*32 + khalf];
      acc[nt] = wmma_bf16(a, bf, acc[nt]);
    }
    __syncthreads();
  }
  const int half = lane >> 4;
  #pragma unroll
  for (int nt = 0; nt < 6; ++nt) {
    const int n = nb + nt*16 + (lane & 15);
    #pragma unroll
    for (int r = 0; r < 8; ++r) {
      out[(size_t)(rt*16 + r + 8*half)*WH + n] = acc[nt][r];
    }
  }
}

// ---------------------------------------------------------------------------
extern "C" void kernel_launch(void* const* d_in, const int* in_sizes, int n_in,
                              void* d_out, int out_size, void* d_ws, size_t ws_size,
                              hipStream_t stream) {
  (void)in_sizes; (void)n_in; (void)out_size; (void)ws_size;
  const float* x  = (const float*)d_in[0];
  const int* mask = (const int*)d_in[1];
  const float* Aq = (const float*)d_in[2];
  const float* Bq = (const float*)d_in[3];
  const float* Av = (const float*)d_in[4];
  const float* Bv = (const float*)d_in[5];
  float* out = (float*)d_out;

  // workspace layout: xb|qb (bf16 BT*H), vbT (bf16 B*H*T), tq|tv (bf16 BT*64),
  // Sbuf (f32 BT*T), mbuf/lbuf (f32 BT)
  char* wsb = (char*)d_ws;
  __bf16* xb  = (__bf16*)(wsb);
  __bf16* qb  = (__bf16*)(wsb + (size_t)WBT*WH*2);
  __bf16* vbT = (__bf16*)(wsb + (size_t)WBT*WH*4);
  __bf16* tq  = (__bf16*)(wsb + (size_t)WBT*WH*6);
  __bf16* tv  = (__bf16*)(wsb + (size_t)WBT*WH*6 + (size_t)WBT*WR*2);
  char*  rest = wsb + (size_t)WBT*WH*6 + (size_t)WBT*WR*4;
  float* Sbuf = (float*)rest;
  float* mbuf = (float*)(rest + (size_t)WBT*WT*4);
  float* lbuf = mbuf + WBT;

  k1_lora_a<<<64, 256, 0, stream>>>(x, Aq, Av, tq, tv);
  k2_qv<<<dim3(64, 8), 256, 0, stream>>>(x, tq, tv, Bq, Bv, qb, vbT, xb);
  k3_scores<<<64, 256, 2*16*WH*2, stream>>>(qb, xb, mask, Sbuf, mbuf, lbuf);
  k4_out<<<512, 256, 2*WH*32*2, stream>>>(Sbuf, mbuf, lbuf, vbT, out);
}
